// KAN_Decode_34351148433545
// MI455X (gfx1250) — matrix operand order repaired and checked by reading, not compiled, not run on hardware
//
#include <hip/hip_runtime.h>
#include <stdint.h>
#include <stddef.h>

// ---------------------------------------------------------------------------
// KAN forward for MI455X (gfx1250, wave32, WMMA bf16 -> f32 accumulate)
//
//   h1 = silu(x @ dense_w.T + dense_b)                  (16384x64)->(16384x512)
//   h2 = kan_linear(h1, W1)   K = 512*9 = 4608          -> (16384x512)
//   out= kan_linear(h2, W2)   K = 4608                  -> (16384x256)
//
// Each KAN layer is ONE bf16 GEMM: per input i the K dimension concatenates
// [silu(h), spline bases g=0..7] against [base_w[o,i], spline_w[o,i,g]*scaler].
// Activation planes are generated on the fly into LDS; uniform-grid cubic
// B-splines have only 4 non-zero bases per x -> closed-form polynomials
// scattered into LDS (no full Cox-de Boor recursion on the VALU).
// ---------------------------------------------------------------------------

typedef __bf16 bf16_t;
typedef __attribute__((ext_vector_type(16))) __bf16 v16bf;
typedef __attribute__((ext_vector_type(8)))  __bf16 v8bf;
typedef __attribute__((ext_vector_type(8)))  float  v8f;

#define BATCH   16384
#define LATENT  64
#define HID     512
#define OUTD    256
#define NB      8                // spline bases per input
#define NPLANE  9                // 1 (silu) + 8 (bases)
#define KC      (HID * NPLANE)   // 4608 combined K

// round-to-nearest-even f32 -> bf16 (pure bit ops)
__device__ __forceinline__ bf16_t f2bf(float f) {
    union { float f; uint32_t u; } c; c.f = f;
    uint32_t r = c.u + 0x7FFFu + ((c.u >> 16) & 1u);
    unsigned short h = (unsigned short)(r >> 16);
    return __builtin_bit_cast(bf16_t, h);
}

__device__ __forceinline__ float silu_f(float x) {
    return x / (1.0f + __expf(-x));
}

// ---------------------------------------------------------------------------
// Prep kernels (tiny, one-time per launch)
// ---------------------------------------------------------------------------
__global__ void cvt_f32_bf16(const float* __restrict__ in, bf16_t* __restrict__ out, int n) {
    int i = blockIdx.x * 256 + threadIdx.x;
    if (i < n) out[i] = f2bf(in[i]);
}

// Combined weight, row-major [O][KC], K layout = plane-major: k = g*512 + i
__global__ void build_wc(const float* __restrict__ base_w,
                         const float* __restrict__ spline_w,
                         const float* __restrict__ scaler,
                         bf16_t* __restrict__ wc, int O) {
    int idx = blockIdx.x * 256 + threadIdx.x;   // o*512 + i
    if (idx >= O * HID) return;
    int o = idx >> 9;
    int i = idx & 511;
    size_t row = (size_t)o * KC;
    wc[row + i] = f2bf(base_w[idx]);
    float sc = scaler[idx];
#pragma unroll
    for (int g = 0; g < NB; ++g)
        wc[row + (size_t)(g + 1) * HID + i] = f2bf(spline_w[(size_t)idx * NB + g] * sc);
}

// ---------------------------------------------------------------------------
// Prelude: h1 = silu(xb @ dwb.T + bias), K = 64 (2 WMMA K-steps)
// block = 256 thr (8 waves, 4 row x 2 col), tile 64 x 128, wave tile 16 x 64
// ---------------------------------------------------------------------------
__global__ __launch_bounds__(256)
void prelude_gemm(const bf16_t* __restrict__ xb, const bf16_t* __restrict__ dwb,
                  const float* __restrict__ bias, float* __restrict__ h1) {
    const int tid   = threadIdx.x;
    const int lane  = tid & 31;
    const int wave  = tid >> 5;
    const int m0    = (wave >> 1) * 16;
    const int row0  = blockIdx.x * 64;
    const int col0  = blockIdx.y * 128 + (wave & 1) * 64;
    const int lhalf = lane >> 4;     // 0|1
    const int lmod  = lane & 15;

    v8f acc[4];
#pragma unroll
    for (int t = 0; t < 4; ++t) acc[t] = (v8f)(0.0f);

#pragma unroll
    for (int ks = 0; ks < LATENT; ks += 32) {
        // A fragment 16x32 bf16 (contiguous k order: k = 16*half + s): two contiguous 16B chunks
        const bf16_t* ap = xb + (size_t)(row0 + m0 + lmod) * LATENT + ks + lhalf * 16;
        v8bf alo = *(const v8bf*)ap;
        v8bf ahi = *(const v8bf*)(ap + 8);
        v16bf a;
#pragma unroll
        for (int i = 0; i < 8; ++i) { a[i] = alo[i]; a[i + 8] = ahi[i]; }
#pragma unroll
        for (int t = 0; t < 4; ++t) {
            int o = col0 + t * 16 + lmod;
            const bf16_t* bp = dwb + (size_t)o * LATENT + ks + lhalf * 16;
            v16bf bfr = *(const v16bf*)bp;
            acc[t] = __builtin_amdgcn_wmma_f32_16x16x32_bf16(
                         false, a, false, bfr, (short)0, acc[t], false, false);
        }
    }
#pragma unroll
    for (int t = 0; t < 4; ++t) {
        int col = col0 + t * 16 + lmod;
        float bv = bias[col];
#pragma unroll
        for (int v = 0; v < 8; ++v) {
            int row = row0 + m0 + v + lhalf * 8;
            h1[(size_t)row * HID + col] = silu_f(acc[t][v] + bv);
        }
    }
}

// ---------------------------------------------------------------------------
// Fused KAN linear: out[b,o] = sum_k Ac[b,k] * Wc[o,k], K = 4608.
// block = 256 thr = 8 waves arranged 2 row-groups x 4 col-groups
// block tile 64 rows x 256 cols; wave tile 32 x 64 (MT=2 A-frags, NT=4 tiles)
// -> per K-step: 2 A frags + 4 B frags feed 8 WMMAs (1 global b128 per WMMA)
// LDS = 9 planes x 64 x 32 bf16 = 36 KB
// ---------------------------------------------------------------------------
#define MT 2
#define NT 4
__global__ __launch_bounds__(256)
void kan_fused(const float* __restrict__ hin, const bf16_t* __restrict__ wc,
               float* __restrict__ out, int O) {
    __shared__ bf16_t lds[NPLANE * 64 * 32];   // 36 KB

    const int tid   = threadIdx.x;
    const int lane  = tid & 31;
    const int wave  = tid >> 5;
    const int wrow  = wave >> 2;         // 0..1 (32-row groups)
    const int wcol  = wave & 3;          // 0..3 (64-col groups)
    const int row0  = blockIdx.x * 64;
    const int col0  = blockIdx.y * 256 + wcol * 64;
    const int lhalf = lane >> 4;
    const int lmod  = lane & 15;
    const int pr    = tid >> 2;          // producer row   0..63
    const int pc    = (tid & 3) * 8;     // producer chunk 0,8,16,24

    v8f acc[MT][NT];
#pragma unroll
    for (int mt = 0; mt < MT; ++mt)
#pragma unroll
        for (int t = 0; t < NT; ++t) acc[mt][t] = (v8f)(0.0f);

    const v8bf zero8 = (v8bf)((__bf16)0.0f);

    for (int ic = 0; ic < HID / 32; ++ic) {
        const float* hp = hin + (size_t)(row0 + pr) * HID + ic * 32 + pc;
        float xv[8];
#pragma unroll
        for (int u = 0; u < 8; ++u) xv[u] = hp[u];

        __syncthreads();   // prior iteration's consumers done with LDS

        // zero the 8 spline planes for my slots (plane 0 fully overwritten)
#pragma unroll
        for (int p = 1; p < NPLANE; ++p)
            *(v8bf*)&lds[p * 2048 + pr * 32 + pc] = zero8;

        // plane 0: silu(x), vectorized
        {
            v8bf s;
#pragma unroll
            for (int u = 0; u < 8; ++u) s[u] = f2bf(silu_f(xv[u]));
            *(v8bf*)&lds[pr * 32 + pc] = s;
        }

        // planes 1..8: 4 non-zero uniform cubic B-spline bases per element
#pragma unroll
        for (int u = 0; u < 8; ++u) {
            float x  = xv[u];
            float s  = (x + 2.2f) * 2.5f;         // knot t_j=(j-3)*0.4-1, h=0.4
            float fm = floorf(s);
            int   m  = (int)fm;                   // deg-0 interval index
            if (m >= 0 && m <= 10) {
                float uu  = s - fm;               // local coord in [0,1)
                float w   = 1.0f - uu;
                float u2  = uu * uu;
                float u3  = u2 * uu;
                const float c6 = 1.0f / 6.0f;
                float b0 = w * w * w * c6;
                float b1 = (3.0f * u3 - 6.0f * u2 + 4.0f) * c6;
                float b2 = (-3.0f * u3 + 3.0f * u2 + 3.0f * uu + 1.0f) * c6;
                float b3 = u3 * c6;
                int e  = pr * 32 + pc + u;
                int j0 = m - 3;                   // basis index of b0
                if (j0 >= 0 && j0 <= 7)         lds[(1 + j0) * 2048 + e] = f2bf(b0);
                if (j0 + 1 >= 0 && j0 + 1 <= 7) lds[(2 + j0) * 2048 + e] = f2bf(b1);
                if (j0 + 2 >= 0 && j0 + 2 <= 7) lds[(3 + j0) * 2048 + e] = f2bf(b2);
                if (j0 + 3 >= 0 && j0 + 3 <= 7) lds[(4 + j0) * 2048 + e] = f2bf(b3);
            }
        }
        __syncthreads();

        // ---- consumer: 9 K-steps of 32; MT A-frags x NT B-frags -> 8 WMMAs ----
#pragma unroll
        for (int g = 0; g < NPLANE; ++g) {
            v16bf a[MT];
#pragma unroll
            for (int mt = 0; mt < MT; ++mt) {
                const bf16_t* ap =
                    &lds[g * 2048 + (wrow * 32 + mt * 16 + lmod) * 32 + lhalf * 16];
                v8bf alo = *(const v8bf*)ap;
                v8bf ahi = *(const v8bf*)(ap + 8);
#pragma unroll
                for (int i = 0; i < 8; ++i) { a[mt][i] = alo[i]; a[mt][i + 8] = ahi[i]; }
            }
#pragma unroll
            for (int t = 0; t < NT; ++t) {
                int o = col0 + t * 16 + lmod;
                const bf16_t* bp = wc + (size_t)o * KC + g * HID + ic * 32 + lhalf * 16;
                v16bf bfr = *(const v16bf*)bp;
#pragma unroll
                for (int mt = 0; mt < MT; ++mt)
                    acc[mt][t] = __builtin_amdgcn_wmma_f32_16x16x32_bf16(
                                     false, a[mt], false, bfr, (short)0,
                                     acc[mt][t], false, false);
            }
        }
    }

#pragma unroll
    for (int mt = 0; mt < MT; ++mt)
#pragma unroll
        for (int t = 0; t < NT; ++t) {
            int col = col0 + t * 16 + lmod;
#pragma unroll
            for (int v = 0; v < 8; ++v) {
                int row = row0 + wrow * 32 + mt * 16 + v + lhalf * 8;
                out[(size_t)row * O + col] = acc[mt][t][v];
            }
        }
}

// ---------------------------------------------------------------------------
extern "C" void kernel_launch(void* const* d_in, const int* in_sizes, int n_in,
                              void* d_out, int out_size, void* d_ws, size_t ws_size,
                              hipStream_t stream) {
    (void)in_sizes; (void)n_in; (void)out_size; (void)ws_size;

    const float* x         = (const float*)d_in[0];
    const float* dense_w   = (const float*)d_in[1];
    const float* dense_b   = (const float*)d_in[2];
    const float* base_w1   = (const float*)d_in[3];
    const float* spline_w1 = (const float*)d_in[4];
    const float* scaler1   = (const float*)d_in[5];
    const float* base_w2   = (const float*)d_in[6];
    const float* spline_w2 = (const float*)d_in[7];
    const float* scaler2   = (const float*)d_in[8];
    float* out = (float*)d_out;

    // workspace layout (~76 MB total)
    char* ws = (char*)d_ws;
    size_t off = 0;
    auto take = [&](size_t bytes) {
        char* p = ws + off;
        off += (bytes + 255) & ~(size_t)255;
        return p;
    };
    bf16_t* xb  = (bf16_t*)take((size_t)BATCH * LATENT * 2);
    bf16_t* dwb = (bf16_t*)take((size_t)HID * LATENT * 2);
    bf16_t* w1c = (bf16_t*)take((size_t)HID * KC * 2);
    bf16_t* w2c = (bf16_t*)take((size_t)OUTD * KC * 2);
    float*  h1  = (float*)take((size_t)BATCH * HID * 4);
    float*  h2  = (float*)take((size_t)BATCH * HID * 4);

    // prep
    cvt_f32_bf16<<<(BATCH * LATENT + 255) / 256, 256, 0, stream>>>(x, xb, BATCH * LATENT);
    cvt_f32_bf16<<<(HID * LATENT + 255) / 256, 256, 0, stream>>>(dense_w, dwb, HID * LATENT);
    build_wc<<<(HID * HID + 255) / 256, 256, 0, stream>>>(base_w1, spline_w1, scaler1, w1c, HID);
    build_wc<<<(OUTD * HID + 255) / 256, 256, 0, stream>>>(base_w2, spline_w2, scaler2, w2c, OUTD);

    // prelude GEMM + silu
    prelude_gemm<<<dim3(BATCH / 64, HID / 128), 256, 0, stream>>>(xb, dwb, dense_b, h1);

    // KAN layer 1: 512 -> 512   (block covers 256 cols -> grid.y = 2)
    kan_fused<<<dim3(BATCH / 64, HID / 256), 256, 0, stream>>>(h1, w1c, h2, HID);

    // KAN layer 2: 512 -> 256   (block covers 256 cols -> grid.y = 1)
    kan_fused<<<dim3(BATCH / 64, OUTD / 256), 256, 0, stream>>>(h2, w2c, out, OUTD);
}
